// Encoder_28802050687022
// MI455X (gfx1250) — matrix operand order repaired and checked by reading, not compiled, not run on hardware
//
#include <hip/hip_runtime.h>
#include <hip/hip_bf16.h>

// ---------------------------------------------------------------------------
// CDNA5 WMMA (wave32): A/B = 16 bf16 per lane (16x32 / 32x16), C/D = 8 f32.
// ---------------------------------------------------------------------------
typedef __attribute__((ext_vector_type(16))) __bf16                       bf16x16;
typedef __attribute__((ext_vector_type(8)))  float                        f32x8;
typedef __attribute__((ext_vector_type(16), aligned(16))) unsigned short  u16x16a;

__device__ __forceinline__ unsigned short f2bf(float f) {
    unsigned int u = __float_as_uint(f);
    unsigned int r = u + 0x7FFFu + ((u >> 16) & 1u);   // RNE
    return (unsigned short)(r >> 16);
}
__device__ __forceinline__ float bf2f(unsigned short h) {
    return __uint_as_float((unsigned int)h << 16);
}
__device__ __forceinline__ float sigmoidf_(float x) {
    return 1.0f / (1.0f + __expf(-x));
}
__device__ __forceinline__ f32x8 wmma_bf16(u16x16a au, u16x16a bu, f32x8 acc) {
    return __builtin_amdgcn_wmma_f32_16x16x32_bf16(
        false, __builtin_bit_cast(bf16x16, au),
        false, __builtin_bit_cast(bf16x16, bu),
        (short)0, acc, false, false);
}
// generic source-element -> bf16 bits
__device__ __forceinline__ unsigned short tobf(float v)          { return f2bf(v); }
__device__ __forceinline__ unsigned short tobf(unsigned short v) { return v; }

// ---------------------------------------------------------------------------
// Weight repack: fp32 [Cout][Cin][S] -> bf16 fragment-major
// [Cout/16][NC][lane(32)][16], lane fragment: row lane%16, k = 16*(lane/16) + j (same order as B loads),
// K' = sp*Cin + ci (channel-minor), zero-padded past Cin*S.
// ---------------------------------------------------------------------------
__global__ void repack_frag(const float* __restrict__ src, unsigned short* __restrict__ dst,
                            int Cin, int S, int NC, int n) {
    int i = blockIdx.x * blockDim.x + threadIdx.x;
    if (i >= n) return;
    int j    = i & 15;
    int lane = (i >> 4) & 31;
    int X    = i >> 9;                 // octile*NC + c
    int c    = X % NC;
    int octile = X / NC;
    int m = lane & 15, hi = lane >> 4;
    int kk = c * 32 + (hi ? 16 : 0) + j;
    float v = 0.f;
    if (kk < Cin * S) {
        int sp = kk / Cin, ci = kk % Cin;
        v = src[((size_t)(octile * 16 + m) * Cin + ci) * S + sp];
    }
    dst[i] = f2bf(v);
}

// bf16 -> f32 output conversion
__global__ void cvt_bf16_f32(const unsigned short* __restrict__ s,
                             float* __restrict__ d, int n) {
    int i = blockIdx.x * blockDim.x + threadIdx.x;
    if (i < n) d[i] = bf2f(s[i]);
}

// ---------------------------------------------------------------------------
// 5x5 conv (pad 2) over concat([xt, h or r*h]) via implicit GEMM + WMMA.
// MODE 0: out = sigmoid(conv + b)           (zr buffer, Cout = 2*C1)
// MODE 1: out = (1-z)*h + z*tanh(conv + b)  (fused GRU update, Cout = C1)
// All activations stored as bf16 bits.  NT = oc-tiles per wave (B reuse).
// Block: 256 threads = 8 waves; grid (H*W/16, 1, B).
// ---------------------------------------------------------------------------
template <int MODE, int C0, int C1, int HH, int WW, int NT>
__global__ __launch_bounds__(256)
void convgru5_wmma(const unsigned short* __restrict__ xt,     // [B][C0][HW]
                   const unsigned short* __restrict__ hprev,  // [B][C1][HW] or null
                   const unsigned short* __restrict__ zr,     // [B][2*C1][HW]
                   const unsigned short* __restrict__ wbf,    // frag-major
                   const float* __restrict__ bias,
                   unsigned short* __restrict__ out) {
    constexpr int CIN   = C0 + C1;
    constexpr int COUT  = MODE ? C1 : 2 * C1;
    constexpr int HW    = HH * WW;
    constexpr int PITCH = CIN + 8;
    constexpr int NC    = (25 * CIN) / 32;        // exact for CIN in {128,256}
    extern __shared__ unsigned short lds[];       // [100][PITCH]

    const int b    = blockIdx.z;
    const int tile = blockIdx.x;
    const int y    = (tile * 16) / WW;
    const int x0   = (tile * 16) % WW;

    // ---- stage halo (5 rows x 20 cols x CIN, channel-minor) into LDS ----
    for (int i = threadIdx.x; i < CIN * 100; i += 256) {
        int ci = i / 100, rem = i % 100;
        int row = rem / 20, col = rem % 20;
        int iy = y + row - 2, ix = x0 + col - 2;
        unsigned short v = 0;
        if (iy >= 0 && iy < HH && ix >= 0 && ix < WW) {
            if (ci < C0) {
                v = xt[((size_t)b * C0 + ci) * HW + iy * WW + ix];
            } else {
                int cj = ci - C0;
                unsigned short hb = hprev ? hprev[((size_t)b * C1 + cj) * HW + iy * WW + ix]
                                          : (unsigned short)0;
                if (MODE == 1) {
                    float rv = bf2f(zr[((size_t)b * 2 * C1 + C1 + cj) * HW + iy * WW + ix]);
                    v = f2bf(bf2f(hb) * rv);
                } else {
                    v = hb;
                }
            }
        }
        lds[rem * PITCH + ci] = v;
    }
    __syncthreads();

    const int wave = threadIdx.x >> 5;
    const int lane = threadIdx.x & 31;
    const int octile = wave;                       // grid.y == 1
    if (octile * 16 >= COUT) return;               // wave-uniform

    const int n    = lane & 15;
    const int hi   = lane >> 4;
    const int hi16 = hi ? 16 : 0;

    const u16x16a* ap0 = (const u16x16a*)wbf + (size_t)octile * NC * 32 + lane;
    const u16x16a* ap1 = ap0 + (size_t)8 * NC * 32;   // second tile (oc + 128)

    f32x8 acc[NT];
#pragma unroll
    for (int tt = 0; tt < NT; ++tt) acc[tt] = (f32x8){};

    int ky = 0, kx = 0;
    for (int s = 0; s < 25; ++s) {
        const int rb = (ky * 20 + n + kx) * PITCH + hi16;
        __builtin_prefetch(ap0 + CIN, 0, 1);
#pragma unroll
        for (int cb = 0; cb < CIN; cb += 32) {
            u16x16a bu = *(const u16x16a*)(&lds[rb + cb]);
            u16x16a a0 = *ap0; ap0 += 32;
            acc[0] = wmma_bf16(a0, bu, acc[0]);
            if (NT == 2) {
                u16x16a a1 = *ap1; ap1 += 32;
                acc[1] = wmma_bf16(a1, bu, acc[1]);
            }
        }
        if (++kx == 5) { kx = 0; ++ky; }
    }

    // ---- epilogue: C/D layout VGPR r -> M=r (lanes0-15) / M=8+r ----
    const int pix = y * WW + x0 + n;
#pragma unroll
    for (int tt = 0; tt < NT; ++tt) {
        const int ob = octile * 16 + tt * 128;
#pragma unroll
        for (int r = 0; r < 8; r++) {
            int oc = ob + r + (hi ? 8 : 0);
            float val = acc[tt][r] + bias[oc];
            if (MODE == 0) {
                out[((size_t)b * COUT + oc) * HW + pix] = f2bf(sigmoidf_(val));
            } else {
                float z  = bf2f(zr[((size_t)b * 2 * C1 + oc) * HW + pix]);
                float hp = hprev ? bf2f(hprev[((size_t)b * C1 + oc) * HW + pix]) : 0.f;
                out[((size_t)b * C1 + oc) * HW + pix] =
                    f2bf((1.f - z) * hp + z * tanhf(val));
            }
        }
    }
}

// ---------------------------------------------------------------------------
// 4x4 conv, stride 2, pad 1 via implicit GEMM + WMMA.
// ST = source element type (float for raw input, ushort bf16 otherwise).
// CIN==1 specialized (single zero-padded K-chunk).  Grid (Ho*Wo/16, 1, B).
// ---------------------------------------------------------------------------
template <typename ST, int CIN, int COUT, int HIN, int WIN>
__global__ __launch_bounds__(256)
void conv4s2_wmma(const ST* __restrict__ src, long long srcBStride,
                  const unsigned short* __restrict__ wbf,  // frag-major
                  const float* __restrict__ bias,
                  unsigned short* __restrict__ out) {      // [B][Cout][HWo] bf16
    constexpr int HO = HIN / 2, WO = WIN / 2;
    constexpr int HWI = HIN * WIN, HWO = HO * WO;
    constexpr int PITCH = (CIN == 1) ? 1 : CIN + 8;
    constexpr int NC = (16 * CIN + 31) / 32;               // 1 for CIN==1
    extern __shared__ unsigned short lds[];                // [136][PITCH]

    const int b    = blockIdx.z;
    const int tile = blockIdx.x;
    const int y    = (tile * 16) / WO;
    const int x0   = (tile * 16) % WO;

    for (int i = threadIdx.x; i < CIN * 136; i += 256) {
        int ci = i / 136, rem = i % 136;
        int row = rem / 34, col = rem % 34;
        int iy = 2 * y - 1 + row, ix = 2 * x0 - 1 + col;
        unsigned short v = 0;
        if (iy >= 0 && iy < HIN && ix >= 0 && ix < WIN)
            v = tobf(src[(size_t)b * srcBStride + (size_t)ci * HWI + iy * WIN + ix]);
        lds[rem * PITCH + ci] = v;
    }
    __syncthreads();

    const int wave = threadIdx.x >> 5;
    const int lane = threadIdx.x & 31;
    const int octile = wave;
    if (octile * 16 >= COUT) return;

    const int n    = lane & 15;
    const int hi   = lane >> 4;
    const int hi16 = hi ? 16 : 0;
    const u16x16a* ap = (const u16x16a*)wbf + (size_t)octile * NC * 32 + lane;

    f32x8 acc = {};
    if constexpr (CIN == 1) {
        u16x16a au = *ap;
        u16x16a bu = {};
        if (hi == 0) {                                     // B rows K=0..15
#pragma unroll
            for (int j = 0; j < 16; ++j) {
                int ky = j >> 2, kx = j & 3;
                bu[j] = lds[ky * 34 + 2 * n + kx];
            }
        }
        acc = wmma_bf16(au, bu, acc);
    } else {
        int ky = 0, kx = 0;
        for (int s = 0; s < 16; ++s) {
            const int rb = (ky * 34 + 2 * n + kx) * PITCH + hi16;
            __builtin_prefetch(ap + CIN, 0, 1);
#pragma unroll
            for (int cb = 0; cb < CIN; cb += 32) {
                u16x16a bu = *(const u16x16a*)(&lds[rb + cb]);
                u16x16a au = *ap; ap += 32;
                acc = wmma_bf16(au, bu, acc);
            }
            if (++kx == 4) { kx = 0; ++ky; }
        }
    }

    const int pix = y * WO + x0 + n;
#pragma unroll
    for (int r = 0; r < 8; r++) {
        int oc = octile * 16 + r + (hi ? 8 : 0);
        out[((size_t)b * COUT + oc) * HWO + pix] = f2bf(acc[r] + bias[oc]);
    }
}

// ---------------------------------------------------------------------------
// Host orchestration: 3 stages x 12 timesteps, all on `stream`.
// ---------------------------------------------------------------------------
extern "C" void kernel_launch(void* const* d_in, const int* in_sizes, int n_in,
                              void* d_out, int out_size, void* d_ws, size_t ws_size,
                              hipStream_t stream) {
    (void)in_sizes; (void)n_in; (void)out_size; (void)ws_size;
    constexpr int B = 8, T = 12;

    const float* input   = (const float*)d_in[0];   // [8,12,1,128,128] fp32
    const float* c1_w    = (const float*)d_in[1];
    const float* c1_b    = (const float*)d_in[2];
    const float* g1_zr_w = (const float*)d_in[3];
    const float* g1_zr_b = (const float*)d_in[4];
    const float* g1_h_w  = (const float*)d_in[5];
    const float* g1_h_b  = (const float*)d_in[6];
    const float* c2_w    = (const float*)d_in[7];
    const float* c2_b    = (const float*)d_in[8];
    const float* g2_zr_w = (const float*)d_in[9];
    const float* g2_zr_b = (const float*)d_in[10];
    const float* g2_h_w  = (const float*)d_in[11];
    const float* g2_h_b  = (const float*)d_in[12];
    const float* c3_w    = (const float*)d_in[13];
    const float* c3_b    = (const float*)d_in[14];
    const float* g3_zr_w = (const float*)d_in[15];
    const float* g3_zr_b = (const float*)d_in[16];
    const float* g3_h_w  = (const float*)d_in[17];
    const float* g3_h_b  = (const float*)d_in[18];

    char* ws = (char*)d_ws;
    size_t off = 0;
    auto carve = [&](size_t bytes) -> char* {
        char* p = ws + off;
        off = (off + bytes + 255) & ~(size_t)255;
        return p;
    };
    // fragment-major weight sizes: (Cout/16) * NC * 512 bf16 elements
    const int n_wc1  = (64 / 16)  * 1   * 512;   // CIN=1,  S=16, NC=1 (padded)
    const int n_wg1z = (128 / 16) * 100 * 512;   // CIN=128,S=25
    const int n_wg1h = (64 / 16)  * 100 * 512;
    const int n_wc2  = (128 / 16) * 32  * 512;   // CIN=64, S=16
    const int n_wg2z = (256 / 16) * 200 * 512;   // CIN=256,S=25
    const int n_wg2h = (128 / 16) * 200 * 512;
    const int n_wc3  = (128 / 16) * 64  * 512;   // CIN=128,S=16
    const int n_wg3z = (256 / 16) * 200 * 512;
    const int n_wg3h = (128 / 16) * 200 * 512;

    unsigned short* w_c1  = (unsigned short*)carve((size_t)n_wc1  * 2);
    unsigned short* w_g1z = (unsigned short*)carve((size_t)n_wg1z * 2);
    unsigned short* w_g1h = (unsigned short*)carve((size_t)n_wg1h * 2);
    unsigned short* w_c2  = (unsigned short*)carve((size_t)n_wc2  * 2);
    unsigned short* w_g2z = (unsigned short*)carve((size_t)n_wg2z * 2);
    unsigned short* w_g2h = (unsigned short*)carve((size_t)n_wg2h * 2);
    unsigned short* w_c3  = (unsigned short*)carve((size_t)n_wc3  * 2);
    unsigned short* w_g3z = (unsigned short*)carve((size_t)n_wg3z * 2);
    unsigned short* w_g3h = (unsigned short*)carve((size_t)n_wg3h * 2);

    // bf16 activation buffers
    unsigned short* xt   = (unsigned short*)carve((size_t)2097152 * 2);
    unsigned short* zr   = (unsigned short*)carve((size_t)4194304 * 2);
    unsigned short* seq1 = (unsigned short*)carve((size_t)T * B * 64  * 4096 * 2);
    unsigned short* seq2 = (unsigned short*)carve((size_t)T * B * 128 * 1024 * 2);
    unsigned short* seq3 = (unsigned short*)carve((size_t)T * B * 128 * 256  * 2);

    auto rp = [&](const float* s, unsigned short* d, int Cin, int S, int NC, int n) {
        repack_frag<<<(n + 255) / 256, 256, 0, stream>>>(s, d, Cin, S, NC, n);
    };
    rp(c1_w,    w_c1,  1,   16, 1,   n_wc1);
    rp(g1_zr_w, w_g1z, 128, 25, 100, n_wg1z);
    rp(g1_h_w,  w_g1h, 128, 25, 100, n_wg1h);
    rp(c2_w,    w_c2,  64,  16, 32,  n_wc2);
    rp(g2_zr_w, w_g2z, 256, 25, 200, n_wg2z);
    rp(g2_h_w,  w_g2h, 256, 25, 200, n_wg2h);
    rp(c3_w,    w_c3,  128, 16, 64,  n_wc3);
    rp(g3_zr_w, w_g3z, 256, 25, 200, n_wg3z);
    rp(g3_h_w,  w_g3h, 256, 25, 200, n_wg3h);

    // dynamic LDS sizes (bytes) matching kernel PITCH formulas
    constexpr int lds_c4_1   = 136 * 1 * 2;
    constexpr int lds_c4_64  = 136 * (64 + 8) * 2;
    constexpr int lds_c4_128 = 136 * (128 + 8) * 2;
    constexpr int lds_g128   = 100 * (128 + 8) * 2;
    constexpr int lds_g256   = 100 * (256 + 8) * 2;

    for (int t = 0; t < T; t++) {
        // ===== stage 1: 128x128 -> 64x64, h1=64 =====
        conv4s2_wmma<float, 1, 64, 128, 128><<<dim3(256, 1, B), 256, lds_c4_1, stream>>>(
            input + (size_t)t * 16384, (long long)T * 16384, w_c1, c1_b, xt);
        const unsigned short* h1p = t ? seq1 + (size_t)(t - 1) * B * 64 * 4096 : nullptr;
        convgru5_wmma<0, 64, 64, 64, 64, 1><<<dim3(256, 1, B), 256, lds_g128, stream>>>(
            xt, h1p, nullptr, w_g1z, g1_zr_b, zr);
        convgru5_wmma<1, 64, 64, 64, 64, 1><<<dim3(256, 1, B), 256, lds_g128, stream>>>(
            xt, h1p, zr, w_g1h, g1_h_b, seq1 + (size_t)t * B * 64 * 4096);

        // ===== stage 2: 64x64 -> 32x32, h2=128 =====
        conv4s2_wmma<unsigned short, 64, 128, 64, 64><<<dim3(64, 1, B), 256, lds_c4_64, stream>>>(
            seq1 + (size_t)t * B * 64 * 4096, (long long)64 * 4096, w_c2, c2_b, xt);
        const unsigned short* h2p = t ? seq2 + (size_t)(t - 1) * B * 128 * 1024 : nullptr;
        convgru5_wmma<0, 128, 128, 32, 32, 2><<<dim3(64, 1, B), 256, lds_g256, stream>>>(
            xt, h2p, nullptr, w_g2z, g2_zr_b, zr);
        convgru5_wmma<1, 128, 128, 32, 32, 1><<<dim3(64, 1, B), 256, lds_g256, stream>>>(
            xt, h2p, zr, w_g2h, g2_h_b, seq2 + (size_t)t * B * 128 * 1024);

        // ===== stage 3: 32x32 -> 16x16, h3=128 =====
        conv4s2_wmma<unsigned short, 128, 128, 32, 32><<<dim3(16, 1, B), 256, lds_c4_128, stream>>>(
            seq2 + (size_t)t * B * 128 * 1024, (long long)128 * 1024, w_c3, c3_b, xt);
        const unsigned short* h3p = t ? seq3 + (size_t)(t - 1) * B * 128 * 256 : nullptr;
        convgru5_wmma<0, 128, 128, 16, 16, 2><<<dim3(16, 1, B), 256, lds_g256, stream>>>(
            xt, h3p, nullptr, w_g3z, g3_zr_b, zr);
        convgru5_wmma<1, 128, 128, 16, 16, 1><<<dim3(16, 1, B), 256, lds_g256, stream>>>(
            xt, h3p, zr, w_g3h, g3_h_b, seq3 + (size_t)t * B * 128 * 256);
    }

    // ---- outputs: (state1, state2, state3) concatenated, bf16 -> f32 ----
    float* out = (float*)d_out;
    const int n1 = B * 64 * 4096, n2 = B * 128 * 1024, n3 = B * 128 * 256;
    cvt_bf16_f32<<<(n1 + 255) / 256, 256, 0, stream>>>(
        seq1 + (size_t)(T - 1) * n1, out, n1);
    cvt_bf16_f32<<<(n2 + 255) / 256, 256, 0, stream>>>(
        seq2 + (size_t)(T - 1) * n2, out + n1, n2);
    cvt_bf16_f32<<<(n3 + 255) / 256, 256, 0, stream>>>(
        seq3 + (size_t)(T - 1) * n3, out + n1 + n2, n3);
}
